// SpatialAttention_56384330661919
// MI455X (gfx1250) — matrix operand order repaired and checked by reading, not compiled, not run on hardware
//
#include <hip/hip_runtime.h>
#include <hip/hip_bf16.h>

// Problem constants (B=4, L=4096, C=1024, num_frames=16)
#define BFR   64      // B * num_frames
#define PP    256     // patches per frame
#define HH    16      // heads
#define DD    64      // head dim
#define CC    1024    // embed dim
#define MM    (BFR * PP)   // 16384 rows
#define NQKV  (3 * CC)     // 3072

typedef __attribute__((ext_vector_type(16))) __bf16 v16bf;
typedef __attribute__((ext_vector_type(8)))  float  v8f;
typedef unsigned short u16t;

struct BF16x16 { uint4 lo, hi; };
union V16U { BF16x16 s; v16bf v; };

static __device__ __forceinline__ v16bf pack16(uint4 lo, uint4 hi) {
    V16U u; u.s.lo = lo; u.s.hi = hi; return u.v;
}

static __device__ __forceinline__ v8f zero8() {
    v8f z;
    #pragma unroll
    for (int i = 0; i < 8; ++i) z[i] = 0.0f;
    return z;
}

static __device__ __forceinline__ u16t f2bf(float f) {
    unsigned u = __float_as_uint(f);
    u += 0x7FFFu + ((u >> 16) & 1u);      // round-to-nearest-even
    return (u16t)(u >> 16);
}

// A tile 16x32 (MxK), bf16, row-major source with leading dim ld (elements).
// Lane L: M = L&15, h = L>>4; a[0..7] = K h*8..h*8+7 ; a[8..15] = K 16+h*8..
static __device__ __forceinline__ v16bf load_a(const u16t* base, int ld,
                                               int row0, int k0, int lane) {
    int m = lane & 15, h = lane >> 4;
    const u16t* p = base + (row0 + m) * ld + k0 + h * 8;
    return pack16(*(const uint4*)(p), *(const uint4*)(p + 16));
}

// B tile 32x16 (KxN), columns K-contiguous: source row n holds K run (ld elems).
// Lane L: N = L&15, h = L>>4; b[0..7] = K h*8..h*8+7 ; b[8..15] = K 16+h*8..
static __device__ __forceinline__ v16bf load_b(const u16t* base, int ld,
                                               int col0, int k0, int lane) {
    int n = lane & 15, h = lane >> 4;
    const u16t* p = base + (col0 + n) * ld + k0 + h * 8;
    return pack16(*(const uint4*)(p), *(const uint4*)(p + 16));
}

static __device__ __forceinline__ v8f wmma_bf16(v16bf a, v16bf b, v8f c) {
    return __builtin_amdgcn_wmma_f32_16x16x32_bf16(false, a, false, b,
                                                   (short)0, c, false, false);
}

// ---------------------------------------------------------------- convert f32 -> bf16
__global__ __launch_bounds__(256) void cvt_bf16(const float* __restrict__ in,
                                                u16t* __restrict__ out, int n) {
    int i = blockIdx.x * 256 + threadIdx.x;
    if (i < n) out[i] = f2bf(in[i]);
}

// ---------------------------------------------------------------- QKV projection GEMM
// C[m,n] = sum_k X[m,k] * W[n,k] + b[n] ; scatter into q/k/vT (vT transposed)
__global__ __launch_bounds__(128) void qkv_gemm(const u16t* __restrict__ X,
                                                const u16t* __restrict__ W,
                                                const float* __restrict__ bias,
                                                u16t* __restrict__ qb,
                                                u16t* __restrict__ kb,
                                                u16t* __restrict__ vtb) {
    const int wave = threadIdx.x >> 5;
    const int lane = threadIdx.x & 31;
    const int m0 = blockIdx.x * 64 + (wave >> 1) * 32;
    const int n0 = blockIdx.y * 64 + (wave & 1) * 32;

    v8f acc[2][2];
    acc[0][0] = zero8(); acc[0][1] = zero8();
    acc[1][0] = zero8(); acc[1][1] = zero8();

    for (int k0 = 0; k0 < CC; k0 += 32) {
        v16bf a0 = load_a(X, CC, m0,      k0, lane);
        v16bf a1 = load_a(X, CC, m0 + 16, k0, lane);
        v16bf b0 = load_b(W, CC, n0,      k0, lane);
        v16bf b1 = load_b(W, CC, n0 + 16, k0, lane);
        acc[0][0] = wmma_bf16(a0, b0, acc[0][0]);
        acc[0][1] = wmma_bf16(a0, b1, acc[0][1]);
        acc[1][0] = wmma_bf16(a1, b0, acc[1][0]);
        acc[1][1] = wmma_bf16(a1, b1, acc[1][1]);
    }

    const int col = lane & 15, h = lane >> 4;
    #pragma unroll
    for (int j = 0; j < 2; ++j) {
        int n = n0 + 16 * j + col;
        int which = n >> 10;           // 0=q 1=k 2=v
        int rem   = n & 1023;
        int head  = rem >> 6;
        int d     = rem & 63;
        float bv  = bias[n];
        #pragma unroll
        for (int i = 0; i < 2; ++i) {
            #pragma unroll
            for (int r = 0; r < 8; ++r) {
                int m  = m0 + 16 * i + r + 8 * h;
                int bf = m >> 8;
                int p  = m & 255;
                u16t o = f2bf(acc[i][j][r] + bv);
                if (which == 0)
                    qb[(((bf * HH + head) * PP + p) * DD) + d] = o;
                else if (which == 1)
                    kb[(((bf * HH + head) * PP + p) * DD) + d] = o;
                else
                    vtb[(((bf * HH + head) * DD + d) * PP) + p] = o;
            }
        }
    }
}

// ---------------------------------------------------------------- fused attention
// One block = one (bf,head), 64 query rows; one wave = 16 query rows.
// Full 16x256 score strip kept in registers; softmax via half-wave shuffles.
__global__ __launch_bounds__(128) void attn_fused(const u16t* __restrict__ qb,
                                                  const u16t* __restrict__ kb,
                                                  const u16t* __restrict__ vtb,
                                                  u16t* __restrict__ ao) {
    __shared__ u16t probs[4][16][PP];   // 32 KB

    const int wave = threadIdx.x >> 5;
    const int lane = threadIdx.x & 31;
    const int bfh  = blockIdx.x;        // 0..1023
    const int bf   = bfh >> 4;
    const int head = bfh & 15;
    const int qrow0 = blockIdx.y * 64 + wave * 16;

    const u16t* qp = qb  + bfh * PP * DD;   // [P][D]
    const u16t* kp = kb  + bfh * PP * DD;   // [P][D]
    const u16t* vp = vtb + bfh * DD * PP;   // [D][P]

    // ---- scores = (q @ k^T) * 1/sqrt(D)
    v16bf a0 = load_a(qp, DD, qrow0, 0,  lane);
    v16bf a1 = load_a(qp, DD, qrow0, 32, lane);

    v8f s[16];
    #pragma unroll
    for (int t = 0; t < 16; ++t) {
        v16bf b0 = load_b(kp, DD, t * 16, 0,  lane);
        v16bf b1 = load_b(kp, DD, t * 16, 32, lane);
        v8f c = zero8();
        c = wmma_bf16(a0, b0, c);
        c = wmma_bf16(a1, b1, c);
        #pragma unroll
        for (int e = 0; e < 8; ++e) c[e] = s[t][e] = c[e] * 0.125f;
    }

    // ---- softmax over 256 cols (16 tiles x 16 lanes within half-wave)
    #pragma unroll
    for (int r = 0; r < 8; ++r) {
        float mx = -3.4e38f;
        #pragma unroll
        for (int t = 0; t < 16; ++t) mx = fmaxf(mx, s[t][r]);
        #pragma unroll
        for (int msk = 1; msk < 16; msk <<= 1)
            mx = fmaxf(mx, __shfl_xor(mx, msk, 32));
        float sum = 0.0f;
        #pragma unroll
        for (int t = 0; t < 16; ++t) {
            float e = __expf(s[t][r] - mx);
            s[t][r] = e;
            sum += e;
        }
        #pragma unroll
        for (int msk = 1; msk < 16; msk <<= 1)
            sum += __shfl_xor(sum, msk, 32);
        float inv = 1.0f / sum;
        #pragma unroll
        for (int t = 0; t < 16; ++t) s[t][r] *= inv;
    }

    // ---- stage probs to LDS (C-layout -> memory row-major) as bf16
    {
        const int col = lane & 15, h = lane >> 4;
        #pragma unroll
        for (int t = 0; t < 16; ++t)
            #pragma unroll
            for (int r = 0; r < 8; ++r)
                probs[wave][r + 8 * h][t * 16 + col] = f2bf(s[t][r]);
    }
    __syncthreads();

    // ---- out = probs @ v  (K = 256, 8 wmma steps; B from vT is K-contiguous)
    v8f o[4];
    #pragma unroll
    for (int j = 0; j < 4; ++j) o[j] = zero8();

    const u16t* plds = &probs[wave][0][0];
    for (int kk = 0; kk < 8; ++kk) {
        v16bf pa = load_a(plds, PP, 0, kk * 32, lane);
        #pragma unroll
        for (int j = 0; j < 4; ++j) {
            v16bf bv = load_b(vp, PP, j * 16, kk * 32, lane);
            o[j] = wmma_bf16(pa, bv, o[j]);
        }
    }

    // ---- store as [BF][P][C] bf16 (row m = bf*256+p, col = head*64+d)
    {
        const int col = lane & 15, h = lane >> 4;
        #pragma unroll
        for (int j = 0; j < 4; ++j) {
            int d = j * 16 + col;
            #pragma unroll
            for (int r = 0; r < 8; ++r) {
                int p = qrow0 + r + 8 * h;
                ao[(bf * PP + p) * CC + head * DD + d] = f2bf(o[j][r]);
            }
        }
    }
}

// ---------------------------------------------------------------- output projection
__global__ __launch_bounds__(128) void out_gemm(const u16t* __restrict__ A,
                                                const u16t* __restrict__ W,
                                                const float* __restrict__ bias,
                                                float* __restrict__ out) {
    const int wave = threadIdx.x >> 5;
    const int lane = threadIdx.x & 31;
    const int m0 = blockIdx.x * 64 + (wave >> 1) * 32;
    const int n0 = blockIdx.y * 64 + (wave & 1) * 32;

    v8f acc[2][2];
    acc[0][0] = zero8(); acc[0][1] = zero8();
    acc[1][0] = zero8(); acc[1][1] = zero8();

    for (int k0 = 0; k0 < CC; k0 += 32) {
        v16bf a0 = load_a(A, CC, m0,      k0, lane);
        v16bf a1 = load_a(A, CC, m0 + 16, k0, lane);
        v16bf b0 = load_b(W, CC, n0,      k0, lane);
        v16bf b1 = load_b(W, CC, n0 + 16, k0, lane);
        acc[0][0] = wmma_bf16(a0, b0, acc[0][0]);
        acc[0][1] = wmma_bf16(a0, b1, acc[0][1]);
        acc[1][0] = wmma_bf16(a1, b0, acc[1][0]);
        acc[1][1] = wmma_bf16(a1, b1, acc[1][1]);
    }

    const int col = lane & 15, h = lane >> 4;
    #pragma unroll
    for (int j = 0; j < 2; ++j) {
        int n = n0 + 16 * j + col;
        float bv = bias[n];
        #pragma unroll
        for (int i = 0; i < 2; ++i) {
            #pragma unroll
            for (int r = 0; r < 8; ++r) {
                int m = m0 + 16 * i + r + 8 * h;
                out[m * CC + n] = acc[i][j][r] + bv;
            }
        }
    }
}

extern "C" void kernel_launch(void* const* d_in, const int* in_sizes, int n_in,
                              void* d_out, int out_size, void* d_ws, size_t ws_size,
                              hipStream_t stream) {
    (void)in_sizes; (void)n_in; (void)out_size; (void)ws_size;

    const float* x      = (const float*)d_in[0];
    const float* qkv_w  = (const float*)d_in[1];
    const float* qkv_b  = (const float*)d_in[2];
    const float* out_w  = (const float*)d_in[3];
    const float* out_b  = (const float*)d_in[4];
    float* out = (float*)d_out;

    const int NX  = MM * CC;        // 16,777,216
    const int NW1 = NQKV * CC;      //  3,145,728
    const int NW2 = CC * CC;        //  1,048,576
    const int NQ  = BFR * HH * PP * DD; // 16,777,216

    u16t* xb   = (u16t*)d_ws;
    u16t* wqkv = xb + NX;
    u16t* wout = wqkv + NW1;
    u16t* qb   = wout + NW2;
    u16t* kb   = qb + NQ;
    u16t* vtb  = kb + NQ;
    u16t* ao   = vtb + NQ;

    cvt_bf16<<<(NX  + 255) / 256, 256, 0, stream>>>(x,     xb,   NX);
    cvt_bf16<<<(NW1 + 255) / 256, 256, 0, stream>>>(qkv_w, wqkv, NW1);
    cvt_bf16<<<(NW2 + 255) / 256, 256, 0, stream>>>(out_w, wout, NW2);

    qkv_gemm<<<dim3(MM / 64, NQKV / 64), 128, 0, stream>>>(xb, wqkv, qkv_b,
                                                           qb, kb, vtb);

    attn_fused<<<dim3(BFR * HH, PP / 64), 128, 0, stream>>>(qb, kb, vtb, ao);

    out_gemm<<<dim3(MM / 64, CC / 64), 128, 0, stream>>>(ao, wout, out_b, out);
}
